// CrossAttentionBlock_56384330661944
// MI455X (gfx1250) — matrix operand order repaired and checked by reading, not compiled, not run on hardware
//
#include <hip/hip_runtime.h>
#include <math.h>

// ---------------- types ----------------
typedef __attribute__((ext_vector_type(16))) __bf16 v16bf;
typedef __attribute__((ext_vector_type(8)))  __bf16 v8bf;
typedef __attribute__((ext_vector_type(8)))  float  v8f;

union V16 { v16bf v; v8bf h[2]; };

// ---------------- problem constants ----------------
#define BB 4096
#define KK 32
#define DD 1024
#define NHH 16
#define HDD 64

// =====================================================================
// Weight convert + swizzle: fp32 [Kd,N] row-major -> bf16 fragment-major.
// Tile (kt,nt) covers K rows kt*32..+31, N cols nt*16..+15.
// Storage: out[((kt*(N/16)+nt)*32 + lane)*16 + idx]
//   lane: n = nt*16 + (lane&15), h = lane>>4
//   idx (0..15): k = kt*32 + h*8 + (idx&7) + (idx>>3)*16   (matches 32x16 bf16 B frag)
// =====================================================================
__global__ void wconv_kernel(const float* __restrict__ W, __bf16* __restrict__ out,
                             int Kd, int N) {
    int gid = blockIdx.x * blockDim.x + threadIdx.x;
    int total = Kd * N;
    if (gid >= total) return;
    int idx  = gid & 15;
    int lane = (gid >> 4) & 31;
    int tile = gid >> 9;
    int ntn  = N >> 4;
    int nt   = tile % ntn;
    int kt   = tile / ntn;
    int h    = lane >> 4;
    int n    = (nt << 4) + (lane & 15);
    int k    = (kt << 5) + (h << 3) + (idx & 7) + ((idx >> 3) << 4);
    out[gid] = (__bf16)W[(size_t)k * N + n];
}

// =====================================================================
// LayerNorm rows of fixed width 1024 (fp32 in -> bf16 out). Wave per row.
// =====================================================================
__global__ __launch_bounds__(256)
void ln_kernel(const float* __restrict__ x, const float* __restrict__ g,
               const float* __restrict__ b, __bf16* __restrict__ y, int rows) {
    int wid  = (blockIdx.x * blockDim.x + threadIdx.x) >> 5;
    int lane = threadIdx.x & 31;
    if (wid >= rows) return;
    const float4* row = (const float4*)(x + (size_t)wid * DD);
    float4 vbuf[8];
    float s = 0.f, ss = 0.f;
    #pragma unroll
    for (int c = 0; c < 8; ++c) {
        float4 t = row[c * 32 + lane];
        vbuf[c] = t;
        s  += t.x + t.y + t.z + t.w;
        ss += t.x*t.x + t.y*t.y + t.z*t.z + t.w*t.w;
    }
    #pragma unroll
    for (int m = 16; m >= 1; m >>= 1) { s += __shfl_xor(s, m); ss += __shfl_xor(ss, m); }
    float mu  = s * (1.f / 1024.f);
    float var = ss * (1.f / 1024.f) - mu * mu;
    float rs  = rsqrtf(var + 1e-5f);
    __bf16* yr = y + (size_t)wid * DD;
    #pragma unroll
    for (int c = 0; c < 8; ++c) {
        int base = (c * 32 + lane) * 4;
        float4 t = vbuf[c];
        yr[base+0] = (__bf16)((t.x - mu) * rs * g[base+0] + b[base+0]);
        yr[base+1] = (__bf16)((t.y - mu) * rs * g[base+1] + b[base+1]);
        yr[base+2] = (__bf16)((t.z - mu) * rs * g[base+2] + b[base+2]);
        yr[base+3] = (__bf16)((t.w - mu) * rs * g[base+3] + b[base+3]);
    }
}

// =====================================================================
// Zero float buffer
// =====================================================================
__global__ void zerof_kernel(float* p, int n) {
    int i = blockIdx.x * blockDim.x + threadIdx.x;
    if (i < n) p[i] = 0.f;
}

// =====================================================================
// Generic WMMA GEMM: out = epilogue(A[M,Kd] @ W[Kd,N] + bias)
// A: bf16 row-major. W: bf16 swizzled (see wconv_kernel).
// Block: 256 thr (8 waves). Block tile 128(M)x128(N); wave tile 32x64.
// MODE 0: bf16 store
// MODE 1: bf16 store of val*gvec[col]; atomicAdd row sum-of-squares of val
// MODE 2: fp32 store of val + resid (Wo / W2 with residual)
// MODE 3: bf16 store of exact GELU(val)  (W1)
// =====================================================================
template<int MODE>
__global__ __launch_bounds__(256)
void gemm_wmma_kernel(const __bf16* __restrict__ A, const __bf16* __restrict__ Wsw,
                      const float* __restrict__ bias, void* __restrict__ outp,
                      const float* __restrict__ gvec, float* __restrict__ ssq,
                      const float* __restrict__ resid, int M, int Kd, int N) {
    int lane = threadIdx.x & 31;
    int wid  = threadIdx.x >> 5;
    int h    = lane >> 4;
    int l16  = lane & 15;
    int m0   = blockIdx.y * 128 + (wid & 3) * 32;
    int n0   = blockIdx.x * 128 + (wid >> 2) * 64;
    int ntn  = N >> 4;

    v8f acc[2][4];
    #pragma unroll
    for (int f = 0; f < 2; ++f)
        #pragma unroll
        for (int g = 0; g < 4; ++g)
            #pragma unroll
            for (int i = 0; i < 8; ++i) acc[f][g][i] = 0.f;

    const __bf16* arow0 = A + (size_t)(m0 + l16) * Kd;
    const __bf16* arow1 = A + (size_t)(m0 + 16 + l16) * Kd;
    const int nk = Kd >> 5;

    for (int kt = 0; kt < nk; ++kt) {
        int kbase = (kt << 5) + h * 8;
        V16 a0, a1;
        a0.h[0] = *(const v8bf*)(arow0 + kbase);
        a0.h[1] = *(const v8bf*)(arow0 + kbase + 16);
        a1.h[0] = *(const v8bf*)(arow1 + kbase);
        a1.h[1] = *(const v8bf*)(arow1 + kbase + 16);
        #pragma unroll
        for (int g = 0; g < 4; ++g) {
            size_t nt = (size_t)(n0 >> 4) + g;
            v16bf bfr = *(const v16bf*)(Wsw + (((size_t)kt * ntn + nt) * 32 + lane) * 16);
            acc[0][g] = __builtin_amdgcn_wmma_f32_16x16x32_bf16(
                false, a0.v, false, bfr, (short)0, acc[0][g], false, false);
            acc[1][g] = __builtin_amdgcn_wmma_f32_16x16x32_bf16(
                false, a1.v, false, bfr, (short)0, acc[1][g], false, false);
        }
    }

    float p[2][8] = {};
    #pragma unroll
    for (int f = 0; f < 2; ++f) {
        #pragma unroll
        for (int g = 0; g < 4; ++g) {
            int col = n0 + g * 16 + l16;
            float bi = bias[col];
            #pragma unroll
            for (int r = 0; r < 8; ++r) {
                size_t mrow = (size_t)(m0 + f * 16 + r + 8 * h);
                float val = acc[f][g][r] + bi;
                if (MODE == 0) {
                    ((__bf16*)outp)[mrow * N + col] = (__bf16)val;
                } else if (MODE == 1) {
                    ((__bf16*)outp)[mrow * N + col] = (__bf16)(val * gvec[col]);
                    p[f][r] += val * val;
                } else if (MODE == 2) {
                    ((float*)outp)[mrow * N + col] = val + resid[mrow * N + col];
                } else { // MODE 3: exact GELU
                    float ge = 0.5f * val * (1.f + erff(val * 0.70710678118654752f));
                    ((__bf16*)outp)[mrow * N + col] = (__bf16)ge;
                }
            }
        }
    }
    if (MODE == 1) {
        #pragma unroll
        for (int f = 0; f < 2; ++f)
            #pragma unroll
            for (int r = 0; r < 8; ++r) {
                float v = p[f][r];
                #pragma unroll
                for (int m = 8; m >= 1; m >>= 1) v += __shfl_xor(v, m);
                if (l16 == 0)
                    atomicAdd(&ssq[m0 + f * 16 + r + 8 * h], v);
            }
    }
}

// =====================================================================
// Fused K+V projection GEMM: one pass over A (kv_in), two weight sets.
//   K path: out = (A@Wk + bk);  store (val*gk) bf16, atomicAdd ssq(val)
//   V path: out = (A@Wv + bv);  store bf16
// Halves HBM traffic on the biggest activation stream and doubles WMMA
// work per loaded A fragment (16 wmma per iter on 4 A-loads + 8 B-loads).
// =====================================================================
__global__ __launch_bounds__(256)
void gemm_kv_kernel(const __bf16* __restrict__ A,
                    const __bf16* __restrict__ Wks, const __bf16* __restrict__ Wvs,
                    const float* __restrict__ bk, const float* __restrict__ bv,
                    const float* __restrict__ gk, float* __restrict__ kssq,
                    __bf16* __restrict__ outk, __bf16* __restrict__ outv,
                    int M, int Kd, int N) {
    int lane = threadIdx.x & 31;
    int wid  = threadIdx.x >> 5;
    int h    = lane >> 4;
    int l16  = lane & 15;
    int m0   = blockIdx.y * 128 + (wid & 3) * 32;
    int n0   = blockIdx.x * 128 + (wid >> 2) * 64;
    int ntn  = N >> 4;

    v8f acck[2][4], accv[2][4];
    #pragma unroll
    for (int f = 0; f < 2; ++f)
        #pragma unroll
        for (int g = 0; g < 4; ++g)
            #pragma unroll
            for (int i = 0; i < 8; ++i) { acck[f][g][i] = 0.f; accv[f][g][i] = 0.f; }

    const __bf16* arow0 = A + (size_t)(m0 + l16) * Kd;
    const __bf16* arow1 = A + (size_t)(m0 + 16 + l16) * Kd;
    const int nk = Kd >> 5;

    for (int kt = 0; kt < nk; ++kt) {
        int kbase = (kt << 5) + h * 8;
        V16 a0, a1;
        a0.h[0] = *(const v8bf*)(arow0 + kbase);
        a0.h[1] = *(const v8bf*)(arow0 + kbase + 16);
        a1.h[0] = *(const v8bf*)(arow1 + kbase);
        a1.h[1] = *(const v8bf*)(arow1 + kbase + 16);
        #pragma unroll
        for (int g = 0; g < 4; ++g) {
            size_t toff = (((size_t)kt * ntn + ((n0 >> 4) + g)) * 32 + lane) * 16;
            v16bf bkf = *(const v16bf*)(Wks + toff);
            acck[0][g] = __builtin_amdgcn_wmma_f32_16x16x32_bf16(
                false, a0.v, false, bkf, (short)0, acck[0][g], false, false);
            acck[1][g] = __builtin_amdgcn_wmma_f32_16x16x32_bf16(
                false, a1.v, false, bkf, (short)0, acck[1][g], false, false);
            v16bf bvf = *(const v16bf*)(Wvs + toff);
            accv[0][g] = __builtin_amdgcn_wmma_f32_16x16x32_bf16(
                false, a0.v, false, bvf, (short)0, accv[0][g], false, false);
            accv[1][g] = __builtin_amdgcn_wmma_f32_16x16x32_bf16(
                false, a1.v, false, bvf, (short)0, accv[1][g], false, false);
        }
    }

    float p[2][8] = {};
    #pragma unroll
    for (int f = 0; f < 2; ++f) {
        #pragma unroll
        for (int g = 0; g < 4; ++g) {
            int col = n0 + g * 16 + l16;
            float bik = bk[col];
            float biv = bv[col];
            float gc  = gk[col];
            #pragma unroll
            for (int r = 0; r < 8; ++r) {
                size_t mrow = (size_t)(m0 + f * 16 + r + 8 * h);
                float vk = acck[f][g][r] + bik;
                outk[mrow * N + col] = (__bf16)(vk * gc);
                p[f][r] += vk * vk;
                outv[mrow * N + col] = (__bf16)(accv[f][g][r] + biv);
            }
        }
    }
    #pragma unroll
    for (int f = 0; f < 2; ++f)
        #pragma unroll
        for (int r = 0; r < 8; ++r) {
            float v = p[f][r];
            #pragma unroll
            for (int m = 8; m >= 1; m >>= 1) v += __shfl_xor(v, m);
            if (l16 == 0)
                atomicAdd(&kssq[m0 + f * 16 + r + 8 * h], v);
        }
}

// =====================================================================
// Attention: one wave per (b, head). Handles the reference's raw-reshape
// head slicing: head row (h,j) = flat k offset h*2048 + j*64, and the
// transposed output scatter out[b, hd*NH + h].
// scores folded rmsnorm: s = (q'.k') * rq * rk / sqrt(HD)
// =====================================================================
__global__ __launch_bounds__(256)
void attn_kernel(const __bf16* __restrict__ qp, const __bf16* __restrict__ kp,
                 const __bf16* __restrict__ vp, const float* __restrict__ qssq,
                 const float* __restrict__ kssq, __bf16* __restrict__ attnout) {
    int gw   = (blockIdx.x * blockDim.x + threadIdx.x) >> 5;
    int lane = threadIdx.x & 31;
    int b    = gw >> 4;      // NH = 16
    int head = gw & 15;

    const __bf16* qrow  = qp + (size_t)b * DD + head * HDD;
    const __bf16* kbase = kp + (size_t)b * (KK * DD) + head * (2 * DD);
    const __bf16* vbase = vp + (size_t)b * (KK * DD) + head * (2 * DD);
    const __bf16* krow  = kbase + lane * HDD;

    float s = 0.f;
    #pragma unroll
    for (int c = 0; c < 8; ++c) {
        v8bf kq = *(const v8bf*)(krow + c * 8);
        v8bf qq = *(const v8bf*)(qrow + c * 8);
        #pragma unroll
        for (int e = 0; e < 8; ++e) s += (float)qq[e] * (float)kq[e];
    }
    float rq = rsqrtf(qssq[b] * (1.f / 1024.f) + 1e-6f);
    int   kk = head * 2 + (lane >> 4);           // original k row index for this head row
    float rk = rsqrtf(kssq[b * KK + kk] * (1.f / 1024.f) + 1e-6f);
    s *= rq * rk * 0.125f;                       // 1/sqrt(64)

    float mx = s;
    #pragma unroll
    for (int m = 16; m >= 1; m >>= 1) mx = fmaxf(mx, __shfl_xor(mx, m));
    float e = __expf(s - mx);
    float sum = e;
    #pragma unroll
    for (int m = 16; m >= 1; m >>= 1) sum += __shfl_xor(sum, m);
    float attn = e / sum;

    float a0 = 0.f, a1 = 0.f;
    #pragma unroll
    for (int j = 0; j < 32; ++j) {
        float aj = __shfl(attn, j);
        const __bf16* vr = vbase + j * HDD;
        a0 += aj * (float)vr[lane];
        a1 += aj * (float)vr[lane + 32];
    }
    // out[b, hd*NH + head]
    attnout[(size_t)b * DD + (size_t)lane * NHH + head]        = (__bf16)a0;
    attnout[(size_t)b * DD + (size_t)(lane + 32) * NHH + head] = (__bf16)a1;
}

// =====================================================================
// Host side
// =====================================================================
extern "C" void kernel_launch(void* const* d_in, const int* in_sizes, int n_in,
                              void* d_out, int out_size, void* d_ws, size_t ws_size,
                              hipStream_t stream) {
    const float* query = (const float*)d_in[0];
    const float* pf    = (const float*)d_in[1];
    const float* lnq_g = (const float*)d_in[2];
    const float* lnq_b = (const float*)d_in[3];
    const float* lnkv_g= (const float*)d_in[4];
    const float* lnkv_b= (const float*)d_in[5];
    const float* Wq    = (const float*)d_in[6];
    const float* bq    = (const float*)d_in[7];
    const float* Wk    = (const float*)d_in[8];
    const float* bk    = (const float*)d_in[9];
    const float* Wv    = (const float*)d_in[10];
    const float* bv    = (const float*)d_in[11];
    const float* rmsq_g= (const float*)d_in[12];
    const float* rmsk_g= (const float*)d_in[13];
    const float* Wo    = (const float*)d_in[14];
    const float* bo    = (const float*)d_in[15];
    const float* mlpn_g= (const float*)d_in[16];
    const float* mlpn_b= (const float*)d_in[17];
    const float* W1    = (const float*)d_in[18];
    const float* b1    = (const float*)d_in[19];
    const float* W2    = (const float*)d_in[20];
    const float* b2    = (const float*)d_in[21];
    float* out = (float*)d_out;

    // ---- workspace layout (bytes) ----
    char* ws = (char*)d_ws;
    size_t off = 0;
    auto alloc = [&](size_t bytes) { size_t o = off; off += (bytes + 255) & ~(size_t)255; return o; };
    const size_t ND   = (size_t)BB * DD;         // 4M elems
    const size_t NKD  = (size_t)BB * KK * DD;    // 128M elems
    size_t o_kvin = alloc(NKD * 2);
    size_t o_qin  = alloc(ND * 2);
    size_t o_wq   = alloc((size_t)DD * DD * 2);
    size_t o_wk   = alloc((size_t)DD * DD * 2);
    size_t o_wv   = alloc((size_t)DD * DD * 2);
    size_t o_wo   = alloc((size_t)DD * DD * 2);
    size_t o_w1   = alloc((size_t)DD * 4 * DD * 2);
    size_t o_w2   = alloc((size_t)DD * 4 * DD * 2);
    size_t o_qp   = alloc(ND * 2);
    size_t o_kp   = alloc(NKD * 2);
    size_t o_v    = alloc(NKD * 2);
    size_t o_qssq = alloc((size_t)BB * 4);
    size_t o_kssq = alloc((size_t)BB * KK * 4);
    size_t o_ao   = alloc(ND * 2);
    size_t o_out1 = alloc(ND * 4);
    size_t o_h    = alloc(ND * 2);
    size_t o_m    = alloc((size_t)BB * 4 * DD * 2);
    (void)ws_size;

    __bf16* kvin = (__bf16*)(ws + o_kvin);
    __bf16* qin  = (__bf16*)(ws + o_qin);
    __bf16* wq   = (__bf16*)(ws + o_wq);
    __bf16* wk   = (__bf16*)(ws + o_wk);
    __bf16* wv   = (__bf16*)(ws + o_wv);
    __bf16* wo   = (__bf16*)(ws + o_wo);
    __bf16* w1   = (__bf16*)(ws + o_w1);
    __bf16* w2   = (__bf16*)(ws + o_w2);
    __bf16* qp   = (__bf16*)(ws + o_qp);
    __bf16* kp   = (__bf16*)(ws + o_kp);
    __bf16* vv   = (__bf16*)(ws + o_v);
    float*  qssq = (float*)(ws + o_qssq);
    float*  kssq = (float*)(ws + o_kssq);
    __bf16* ao   = (__bf16*)(ws + o_ao);
    float*  out1 = (float*)(ws + o_out1);
    __bf16* hbuf = (__bf16*)(ws + o_h);
    __bf16* mbuf = (__bf16*)(ws + o_m);

    // ---- 1. weight convert/swizzle ----
    {
        int tdd = DD * DD, t14 = DD * 4 * DD;
        wconv_kernel<<<(tdd + 255) / 256, 256, 0, stream>>>(Wq, wq, DD, DD);
        wconv_kernel<<<(tdd + 255) / 256, 256, 0, stream>>>(Wk, wk, DD, DD);
        wconv_kernel<<<(tdd + 255) / 256, 256, 0, stream>>>(Wv, wv, DD, DD);
        wconv_kernel<<<(tdd + 255) / 256, 256, 0, stream>>>(Wo, wo, DD, DD);
        wconv_kernel<<<(t14 + 255) / 256, 256, 0, stream>>>(W1, w1, DD, 4 * DD);
        wconv_kernel<<<(t14 + 255) / 256, 256, 0, stream>>>(W2, w2, 4 * DD, DD);
    }
    // ---- 2. zero ssq accumulators (atomics; must be zeroed every call) ----
    {
        int n = BB + BB * KK;   // qssq and kssq are contiguous
        zerof_kernel<<<(n + 255) / 256, 256, 0, stream>>>(qssq, n);
    }
    // ---- 3. layernorms -> bf16 ----
    ln_kernel<<<BB / 8, 256, 0, stream>>>(query, lnq_g, lnq_b, qin, BB);
    ln_kernel<<<(BB * KK) / 8, 256, 0, stream>>>(pf, lnkv_g, lnkv_b, kvin, BB * KK);

    // ---- 4. projections ----
    gemm_wmma_kernel<1><<<dim3(DD / 128, BB / 128), 256, 0, stream>>>(
        qin, wq, bq, qp, rmsq_g, qssq, nullptr, BB, DD, DD);
    gemm_kv_kernel<<<dim3(DD / 128, (BB * KK) / 128), 256, 0, stream>>>(
        kvin, wk, wv, bk, bv, rmsk_g, kssq, kp, vv, BB * KK, DD, DD);

    // ---- 5. attention (wave per (b,head)) ----
    attn_kernel<<<(BB * NHH) / 8, 256, 0, stream>>>(qp, kp, vv, qssq, kssq, ao);

    // ---- 6. output projection + residual -> out1 (fp32) ----
    gemm_wmma_kernel<2><<<dim3(DD / 128, BB / 128), 256, 0, stream>>>(
        ao, wo, bo, out1, nullptr, nullptr, query, BB, DD, DD);

    // ---- 7. MLP ----
    ln_kernel<<<BB / 8, 256, 0, stream>>>(out1, mlpn_g, mlpn_b, hbuf, BB);
    gemm_wmma_kernel<3><<<dim3((4 * DD) / 128, BB / 128), 256, 0, stream>>>(
        hbuf, w1, b1, mbuf, nullptr, nullptr, nullptr, BB, DD, 4 * DD);
    gemm_wmma_kernel<2><<<dim3(DD / 128, BB / 128), 256, 0, stream>>>(
        mbuf, w2, b2, out, nullptr, nullptr, out1, BB, 4 * DD, DD);
    (void)in_sizes; (void)n_in; (void)out_size;
}
